// KeypointDetectorSelfAttention_69630009803392
// MI455X (gfx1250) — matrix operand order repaired and checked by reading, not compiled, not run on hardware
//
#include <hip/hip_runtime.h>
#include <hip/hip_bf16.h>
#include <math.h>

typedef __attribute__((ext_vector_type(16))) _Float16 v16h;
typedef __attribute__((ext_vector_type(8)))  _Float16 v8h;
typedef __attribute__((ext_vector_type(8)))  float    v8f;

#define BB 2
#define NN 16384
#define MM 1024
#define KK 32
#define CC 64
#define EPSF 1e-5f

// ---------------------------------------------------------------------------
// KNN: one block per (b,m). Each of 128 threads keeps a private sorted top-32
// over its strided slice of the 16384 points, then candidates are merged in
// LDS with 32 rounds of block-wide argmin (wave32 shfl reduction).
// ---------------------------------------------------------------------------
__global__ __launch_bounds__(128) void knn_kernel(const float* __restrict__ xyz,
                                                  const int* __restrict__ rand_idx,
                                                  int* __restrict__ knn_idx) {
  const int bm = blockIdx.x;
  const int b = bm / MM, m = bm % MM;
  const int tid = threadIdx.x;
  const float* xb = xyz + (size_t)b * NN * 3;
  const int sidx = rand_idx[m];
  const float sx = xb[sidx * 3 + 0], sy = xb[sidx * 3 + 1], sz = xb[sidx * 3 + 2];

  float kd[KK]; int ki[KK];
#pragma unroll
  for (int i = 0; i < KK; ++i) { kd[i] = 3.4e38f; ki[i] = 0; }
  for (int n = tid; n < NN; n += 128) {
    float dx = xb[n * 3 + 0] - sx, dy = xb[n * 3 + 1] - sy, dz = xb[n * 3 + 2] - sz;
    float d = dx * dx + dy * dy + dz * dz;
    if (d < kd[KK - 1]) {
      int p = KK - 1;
      while (p > 0 && kd[p - 1] > d) { kd[p] = kd[p - 1]; ki[p] = ki[p - 1]; --p; }
      kd[p] = d; ki[p] = n;
    }
  }
  __shared__ float cd[128 * KK];
  __shared__ int   ci[128 * KK];
  __shared__ float rv[4];
  __shared__ int   rp[4];
#pragma unroll
  for (int i = 0; i < KK; ++i) { cd[tid * KK + i] = kd[i]; ci[tid * KK + i] = ki[i]; }
  __syncthreads();
  const int lane = tid & 31, wid = tid >> 5;
  for (int r = 0; r < KK; ++r) {
    float v = 3.4e38f; int p = 0;
    for (int e = tid; e < 128 * KK; e += 128) {
      float c = cd[e];
      if (c < v) { v = c; p = e; }
    }
#pragma unroll
    for (int off = 16; off > 0; off >>= 1) {
      float ov = __shfl_down(v, (unsigned)off, 32);
      int   op = __shfl_down(p, (unsigned)off, 32);
      if (ov < v) { v = ov; p = op; }
    }
    if (lane == 0) { rv[wid] = v; rp[wid] = p; }
    __syncthreads();
    if (tid == 0) {
      float bv = rv[0]; int bp = rp[0];
      for (int w = 1; w < 4; ++w) if (rv[w] < bv) { bv = rv[w]; bp = rp[w]; }
      knn_idx[bm * KK + r] = ci[bp];
      cd[bp] = 3.4e38f;
    }
    __syncthreads();
  }
}

// ---------------------------------------------------------------------------
// Build g = [rela(3), rdist(1), knn_feat(64)] per row, write as f16 padded to
// 96 K-columns (zeros) inside a 256-wide activation buffer.
// ---------------------------------------------------------------------------
__global__ __launch_bounds__(96) void build_g_kernel(const float* __restrict__ xyz,
                                                     const float* __restrict__ features,
                                                     const int* __restrict__ rand_idx,
                                                     const int* __restrict__ knn_idx,
                                                     float* __restrict__ knn_xyz,
                                                     _Float16* __restrict__ X, int ldX) {
  const int r = blockIdx.x;            // 0 .. B*M*K-1
  const int c = threadIdx.x;           // 0 .. 95
  const int b = r / (MM * KK);
  const int m = (r / KK) % MM;
  const int n = knn_idx[r];
  const float* xb = xyz + (size_t)b * NN * 3;
  const int si = rand_idx[m];
  float val = 0.f;
  if (c < 3) {
    float x = xb[n * 3 + c];
    knn_xyz[(size_t)r * 3 + c] = x;
    val = x - xb[si * 3 + c];
  } else if (c == 3) {
    float dx = xb[n * 3 + 0] - xb[si * 3 + 0];
    float dy = xb[n * 3 + 1] - xb[si * 3 + 1];
    float dz = xb[n * 3 + 2] - xb[si * 3 + 2];
    val = sqrtf(dx * dx + dy * dy + dz * dz);
  } else if (c < 4 + CC) {
    val = features[((size_t)b * CC + (c - 4)) * NN + n];
  }
  X[(size_t)r * ldX + c] = (_Float16)val;
}

// fp32 weight (Kin x Nout, row-major) -> f16 transposed (Nout x Kpad), zero pad
__global__ void convert_weight_kernel(const float* __restrict__ W, _Float16* __restrict__ Wt,
                                      int Kin, int Kpad, int Nout) {
  int i = blockIdx.x * blockDim.x + threadIdx.x;
  if (i >= Nout * Kpad) return;
  int n = i / Kpad, k = i % Kpad;
  Wt[i] = (k < Kin) ? (_Float16)W[(size_t)k * Nout + n] : (_Float16)0.f;
}

// ---------------------------------------------------------------------------
// WMMA GEMM: C(RxN) = A(RxK,f16) * Wt^T. Wt stored NxK with ldB == Kdim, so a
// block's 64-column weight panel is one contiguous region of 64*Kdim halves
// (<= 32KB). All 8 waves of a block share the same column tile (rowTiles is a
// multiple of 8), so the panel is staged ONCE per block into LDS with
// CDNA5 async global->LDS copies (ASYNCcnt-tracked, no VGPR round trip),
// then the WMMA loop reads B from LDS while A streams from global.
// A lane layout used here (contiguous k, same order for B; not AMD's table order): lanes 0-15 row=lane, halves = K[kb..kb+15];
// lanes 16-31 row=lane-16, halves = K[kb+16..kb+31].
// ---------------------------------------------------------------------------
__global__ __launch_bounds__(256) void gemm_wmma_kernel(const _Float16* __restrict__ A,
                                                        const _Float16* __restrict__ Bt,
                                                        float* __restrict__ Cf,
                                                        _Float16* __restrict__ Ch,
                                                        int R, int Kdim, int Ncols,
                                                        int ldA, int ldB, int ldC) {
  __shared__ _Float16 Bs[64 * 256];                // up to 32KB weight panel
  const int wave = (blockIdx.x * blockDim.x + threadIdx.x) >> 5;
  const int lane = threadIdx.x & 31;
  const int rowTiles = R >> 4;
  const int rt = wave % rowTiles;
  const int ct = wave / rowTiles;                  // uniform across the block

  // --- stage the B panel: 64*Kdim halves, contiguous (ldB == Kdim) ---
  {
    const _Float16* pb = Bt + (size_t)(ct << 6) * ldB;
    const int chunks = (Kdim << 6) >> 3;           // 16-byte chunks
    for (int i = threadIdx.x; i < chunks; i += 256) {
      // 16-byte global->LDS copy; low 32 bits of a flat shared
      // pointer are the wave-relative LDS byte offset (ISA 10.2).
      asm volatile("global_load_async_to_lds_b128 %0, %1, off"
                   :
                   : "v"((unsigned)(unsigned long long)(Bs + (size_t)i * 8)),
                     "v"(pb + (size_t)i * 8)
                   : "memory");
    }
    asm volatile("s_wait_asynccnt 0x0" ::: "memory");
  }
  __syncthreads();

  const int halfSel = lane >> 4;                   // 0 | 1
  const int l15 = lane & 15;
  const int arow = (rt << 4) + l15;
  const _Float16* arowp = A + (size_t)arow * ldA + (halfSel << 4);

  v8f acc[4] = {};
  for (int kb = 0; kb < Kdim; kb += 32) {
    v16h a = *(const v16h*)(arowp + kb);           // K = kb + 16*halfSel .. +15
    const int kB = kb + (halfSel << 4);
    __builtin_prefetch(arowp + kb + 64, 0, 0);     // global_prefetch_b8
#pragma unroll
    for (int t = 0; t < 4; ++t) {
      const int colLoc = (t << 4) + l15;           // 0..63 within panel
      v16h bv = *(const v16h*)(Bs + (size_t)colLoc * ldB + kB);
      acc[t] = __builtin_amdgcn_wmma_f32_16x16x32_f16(
          false, a, false, bv, (short)0, acc[t], false, false);
    }
  }
  // C/D layout: lanes 0-15 hold N=lane, M=vgpr; lanes 16-31 hold N=lane-16, M=vgpr+8
  const int rowbase = rt << 4;
#pragma unroll
  for (int t = 0; t < 4; ++t) {
    const int col = (ct << 6) + (t << 4) + l15;
#pragma unroll
    for (int r8 = 0; r8 < 8; ++r8) {
      const int mrow = (halfSel << 3) + r8;
      float v = acc[t][r8];
      if (Ch) Ch[(size_t)(rowbase + mrow) * ldC + col] = (_Float16)v;
      else    Cf[(size_t)(rowbase + mrow) * ldC + col] = v;
    }
  }
}

// Per-column mean / inv-std over R rows (population variance, like jnp.var).
__global__ __launch_bounds__(256) void col_stats_kernel(const float* __restrict__ Y,
                                                        float* __restrict__ stats,
                                                        int R, int ldY) {
  const int c = blockIdx.x;
  float s = 0.f, ss = 0.f;
  for (int r = threadIdx.x; r < R; r += 256) {
    float v = Y[(size_t)r * ldY + c];
    s += v; ss += v * v;
  }
  __shared__ float sh[256], sh2[256];
  sh[threadIdx.x] = s; sh2[threadIdx.x] = ss;
  __syncthreads();
  for (int o = 128; o > 0; o >>= 1) {
    if (threadIdx.x < o) { sh[threadIdx.x] += sh[threadIdx.x + o]; sh2[threadIdx.x] += sh2[threadIdx.x + o]; }
    __syncthreads();
  }
  if (threadIdx.x == 0) {
    float mean = sh[0] / (float)R;
    float var  = sh2[0] / (float)R - mean * mean;
    stats[2 * c]     = mean;
    stats[2 * c + 1] = rsqrtf(var + EPSF);
  }
}

// Normalize + gamma/beta + ReLU, write f16 activations for the next WMMA stage.
__global__ __launch_bounds__(256) void bn_relu_kernel(const float* __restrict__ Y,
                                                      _Float16* __restrict__ X,
                                                      const float* __restrict__ gamma,
                                                      const float* __restrict__ beta,
                                                      const float* __restrict__ stats,
                                                      int R, int Ncols, int ldY, int ldX) {
  size_t i = (size_t)blockIdx.x * 256 + threadIdx.x;
  if (i >= (size_t)R * Ncols) return;
  int r = (int)(i / Ncols), c = (int)(i % Ncols);
  float v = (Y[(size_t)r * ldY + c] - stats[2 * c]) * stats[2 * c + 1] * gamma[c] + beta[c];
  X[(size_t)r * ldX + c] = (_Float16)fmaxf(v, 0.f);
}

// ---------------------------------------------------------------------------
// Attention per (b,m) group of K=32 rows. Uses the identity:
//   att_feat = (sum_q softmax_row)^T @ V ,  keypoints = same colsum @ knn_xyz
// ---------------------------------------------------------------------------
__global__ __launch_bounds__(256) void attention_kernel(const _Float16* __restrict__ QKV,
                                                        const float* __restrict__ knn_xyz,
                                                        _Float16* __restrict__ att16,
                                                        float* __restrict__ out_kp,
                                                        float* __restrict__ out_att) {
  const int bm = blockIdx.x;
  const int b = bm / MM, m = bm % MM;
  const int tid = threadIdx.x;
  const size_t r0 = (size_t)bm * KK;
  __shared__ float Qs[KK * 64];
  __shared__ float Ks[KK * 64];
  __shared__ float Sc[KK * KK];
  __shared__ float Cs[KK];
#pragma unroll
  for (int k = 0; k < 8; ++k) {
    int i = tid + k * 256;
    int j = i >> 6, c = i & 63;
    Qs[i] = (float)QKV[(r0 + j) * 384 + c];
    Ks[i] = (float)QKV[(r0 + j) * 384 + 64 + c];
  }
  __syncthreads();
  const float scale = 0.17677669529663688f;   // 1/sqrt(K=32)
#pragma unroll
  for (int k = 0; k < 4; ++k) {
    int i = tid + k * 256;
    int q = i >> 5, j = i & 31;
    float s = 0.f;
    for (int c = 0; c < 64; ++c) s += Qs[q * 64 + c] * Ks[j * 64 + c];
    Sc[i] = s * scale;
  }
  __syncthreads();
  if (tid < KK) {                              // softmax over j, one row per lane
    float mx = -3.4e38f;
    for (int j = 0; j < KK; ++j) mx = fmaxf(mx, Sc[tid * KK + j]);
    float sum = 0.f;
    for (int j = 0; j < KK; ++j) { float e = __expf(Sc[tid * KK + j] - mx); Sc[tid * KK + j] = e; sum += e; }
    float inv = 1.f / sum;
    for (int j = 0; j < KK; ++j) Sc[tid * KK + j] *= inv;
  }
  __syncthreads();
  if (tid < KK) {                              // column-sum over q
    float s = 0.f;
    for (int q = 0; q < KK; ++q) s += Sc[q * KK + tid];
    Cs[tid] = s;
  }
  __syncthreads();
  {                                            // att_feat channel tid (0..255)
    float af = 0.f;
    for (int j = 0; j < KK; ++j) af += Cs[j] * (float)QKV[(r0 + j) * 384 + 128 + tid];
    att16[(size_t)bm * 256 + tid] = (_Float16)af;
    out_att[((size_t)b * 256 + tid) * MM + m] = af;   // att_feat.transpose(0,2,1)
  }
  if (tid < 3) {
    float kp = 0.f;
    for (int j = 0; j < KK; ++j) kp += Cs[j] * knn_xyz[(r0 + j) * 3 + tid];
    out_kp[(size_t)bm * 3 + tid] = kp;
  }
}

// sig = softplus(h @ m3w + m3b) + 1e-3, one block per (b,m)
__global__ __launch_bounds__(256) void final_sig_kernel(const _Float16* __restrict__ H,
                                                        const float* __restrict__ m3w,
                                                        const float* __restrict__ m3b,
                                                        float* __restrict__ out_sig) {
  const int bm = blockIdx.x;
  const int tid = threadIdx.x;
  float v = (float)H[(size_t)bm * 256 + tid] * m3w[tid];
  __shared__ float sh[256];
  sh[tid] = v;
  __syncthreads();
  for (int o = 128; o > 0; o >>= 1) {
    if (tid < o) sh[tid] += sh[tid + o];
    __syncthreads();
  }
  if (tid == 0) {
    float x = sh[0] + m3b[0];
    float sp = (x > 20.f) ? x : log1pf(__expf(x));
    out_sig[bm] = sp + 0.001f;
  }
}

// ---------------------------------------------------------------------------
extern "C" void kernel_launch(void* const* d_in, const int* in_sizes, int n_in,
                              void* d_out, int out_size, void* d_ws, size_t ws_size,
                              hipStream_t stream) {
  (void)in_sizes; (void)n_in; (void)out_size; (void)ws_size;
  const float* xyz      = (const float*)d_in[0];
  const float* features = (const float*)d_in[1];
  const int*   rand_idx = (const int*)d_in[2];
  const float* w1  = (const float*)d_in[3];
  const float* g1  = (const float*)d_in[4];
  const float* b1  = (const float*)d_in[5];
  const float* w2  = (const float*)d_in[6];
  const float* g2  = (const float*)d_in[7];
  const float* b2  = (const float*)d_in[8];
  const float* w3  = (const float*)d_in[9];
  const float* g3  = (const float*)d_in[10];
  const float* b3  = (const float*)d_in[11];
  const float* wq  = (const float*)d_in[12];
  const float* wk  = (const float*)d_in[13];
  const float* wv  = (const float*)d_in[14];
  const float* m1w = (const float*)d_in[15];
  const float* m1g = (const float*)d_in[17];
  const float* m1be= (const float*)d_in[18];
  const float* m2w = (const float*)d_in[19];
  const float* m2g = (const float*)d_in[21];
  const float* m2be= (const float*)d_in[22];
  const float* m3w = (const float*)d_in[23];
  const float* m3b = (const float*)d_in[24];

  char* ws = (char*)d_ws;
  size_t off = 0;
  auto wsalloc = [&](size_t bytes) -> char* {
    char* p = ws + off;
    off += (bytes + 255) & ~(size_t)255;
    return p;
  };
  const size_t Rbig = (size_t)BB * MM * KK;                 // 65536 rows
  int*      knn_idx = (int*)      wsalloc(Rbig * 4);
  float*    knn_xyz = (float*)    wsalloc(Rbig * 3 * 4);
  _Float16* XA      = (_Float16*) wsalloc(Rbig * 256 * 2);  // activation ping (ld 256)
  _Float16* XB      = (_Float16*) wsalloc(Rbig * 256 * 2);  // activation pong (ld 256)
  float*    Yf      = (float*)    wsalloc(Rbig * 256 * 4);  // fp32 GEMM output
  _Float16* QKV     = (_Float16*) wsalloc(Rbig * 384 * 2);
  _Float16* att16   = (_Float16*) wsalloc((size_t)BB * MM * 256 * 2);
  float*    stats   = (float*)    wsalloc(512 * 4);
  _Float16* W1t     = (_Float16*) wsalloc(128 * 96 * 2);
  _Float16* W2t     = (_Float16*) wsalloc(128 * 128 * 2);
  _Float16* W3t     = (_Float16*) wsalloc(256 * 128 * 2);
  _Float16* WQKVt   = (_Float16*) wsalloc(384 * 256 * 2);
  _Float16* M1t     = (_Float16*) wsalloc(256 * 256 * 2);
  _Float16* M2t     = (_Float16*) wsalloc(256 * 256 * 2);

  float* out_kp  = (float*)d_out;                    // (B,M,3)
  float* out_sig = out_kp + (size_t)BB * MM * 3;     // (B,M)
  float* out_att = out_sig + (size_t)BB * MM;        // (B,C3,M)

  auto cw = [&](const float* W, _Float16* Wt, int Kin, int Kpad, int Nout) {
    int total = Nout * Kpad;
    convert_weight_kernel<<<(total + 255) / 256, 256, 0, stream>>>(W, Wt, Kin, Kpad, Nout);
  };
  cw(w1, W1t, 68, 96, 128);
  cw(w2, W2t, 128, 128, 128);
  cw(w3, W3t, 128, 128, 256);
  cw(wq, WQKVt + 0 * 256,   256, 256, 64);
  cw(wk, WQKVt + 64 * 256,  256, 256, 64);
  cw(wv, WQKVt + 128 * 256, 256, 256, 256);
  cw(m1w, M1t, 256, 256, 256);
  cw(m2w, M2t, 256, 256, 256);

  knn_kernel<<<BB * MM, 128, 0, stream>>>(xyz, rand_idx, knn_idx);
  build_g_kernel<<<(int)Rbig, 96, 0, stream>>>(xyz, features, rand_idx, knn_idx, knn_xyz, XA, 256);

  auto gemm = [&](const _Float16* A, const _Float16* Bt, float* Cf, _Float16* Ch,
                  int R, int Kd, int Nc, int ldA, int ldB, int ldC) {
    int tiles = (R / 16) * (Nc / 64);   // always an exact multiple of 8 waves
    gemm_wmma_kernel<<<tiles / 8, 256, 0, stream>>>(A, Bt, Cf, Ch, R, Kd, Nc, ldA, ldB, ldC);
  };
  auto stats_bn = [&](float* Y, const float* g, const float* be, _Float16* X,
                      int R, int Nc, int ldY, int ldX) {
    col_stats_kernel<<<Nc, 256, 0, stream>>>(Y, stats, R, ldY);
    size_t total = (size_t)R * Nc;
    bn_relu_kernel<<<(unsigned)((total + 255) / 256), 256, 0, stream>>>(Y, X, g, be, stats, R, Nc, ldY, ldX);
  };

  // MLP chain on 65536 rows (f16 WMMA, f32 accumulate)
  gemm(XA, W1t, Yf, nullptr, 65536, 96, 128, 256, 96, 128);
  stats_bn(Yf, g1, b1, XB, 65536, 128, 128, 256);
  gemm(XB, W2t, Yf, nullptr, 65536, 128, 128, 256, 128, 128);
  stats_bn(Yf, g2, b2, XA, 65536, 128, 128, 256);
  gemm(XA, W3t, Yf, nullptr, 65536, 128, 256, 256, 128, 256);
  stats_bn(Yf, g3, b3, XB, 65536, 256, 256, 256);      // XB = e
  // Fused QKV projection, f16 output
  gemm(XB, WQKVt, nullptr, QKV, 65536, 256, 384, 256, 256, 384);
  attention_kernel<<<BB * MM, 256, 0, stream>>>(QKV, knn_xyz, att16, out_kp, out_att);
  // Head MLP on 2048 rows (biases cancel through BN mean-subtraction)
  gemm(att16, M1t, Yf, nullptr, 2048, 256, 256, 256, 256, 256);
  stats_bn(Yf, m1g, m1be, XA, 2048, 256, 256, 256);
  gemm(XA, M2t, Yf, nullptr, 2048, 256, 256, 256, 256, 256);
  stats_bn(Yf, m2g, m2be, XB, 2048, 256, 256, 256);
  final_sig_kernel<<<BB * MM, 256, 0, stream>>>(XB, m3w, m3b, out_sig);
}
